// Attention_76055280877689
// MI455X (gfx1250) — matrix operand order repaired and checked by reading, not compiled, not run on hardware
//
#include <hip/hip_runtime.h>

// ---------------------------------------------------------------------------
// Causal multi-head attention layer for MI455X (gfx1250), all matmuls via
// v_wmma_f32_16x16x32_bf16 (f32 accumulate), softmax stats in f32.
// ---------------------------------------------------------------------------

#define D_MODEL 1024
#define T_SEQ   4096
#define N_HEADS 16
#define HEAD_DIM 64
#define ATTN_SCALE 0.03125f   // 1024^-0.5
#define NEG_BIG (-1e30f)

typedef __attribute__((ext_vector_type(16))) __bf16          v16bf;
typedef __attribute__((ext_vector_type(8)))  float           v8f;
typedef __attribute__((ext_vector_type(8)))  unsigned short  ushort8;
typedef __attribute__((ext_vector_type(16))) unsigned short  ushort16;

struct U16Pair { ushort8 lo; ushort8 hi; };

static __device__ __forceinline__ unsigned short f2bf(float f) {
    unsigned int u = __float_as_uint(f);
    u += 0x7FFFu + ((u >> 16) & 1u);   // round-to-nearest-even
    return (unsigned short)(u >> 16);
}

static __device__ __forceinline__ v8f wmma_bf16(v16bf a, v16bf b, v8f c) {
    return __builtin_amdgcn_wmma_f32_16x16x32_bf16(
        /*neg_a=*/false, a, /*neg_b=*/false, b,
        /*c_mod=*/(short)0, c, /*reuse_a=*/false, /*reuse_b=*/false);
}

// A-fragment (16x32 bf16, MxK): lane = row (lane&15); lanes 0-15 hold K 0-15,
// lanes 16-31 hold K 16-31 (two 16B contiguous loads).
static __device__ __forceinline__ v16bf load_a_frag(
        const unsigned short* base, int ld, int lane) {
    int row = lane & 15;
    int kb  = (lane & 16) ? 16 : 0;
    const unsigned short* p = base + (size_t)row * ld + kb;
    U16Pair f;
    f.lo = *(const ushort8*)(p);
    f.hi = *(const ushort8*)(p + 8);
    return __builtin_bit_cast(v16bf, f);
}

// B-fragment (32x16 bf16, KxN): lane = column (lane&15); lanes 0-15 hold
// K kbase..kbase+15, lanes 16-31 hold kbase+16..kbase+31 (one 32B load).
// Source is row-major with "rows" = B columns, "cols" = K dimension.
static __device__ __forceinline__ v16bf load_b_frag(
        const unsigned short* src, int ld, int lane, int rowbase, int kbase) {
    int r = rowbase + (lane & 15);
    int k = kbase + ((lane & 16) ? 16 : 0);
    ushort16 u = *(const ushort16*)(src + (size_t)r * ld + k);
    return __builtin_bit_cast(v16bf, u);
}

// ---------------------------------------------------------------------------
// f32 -> bf16 conversion
// ---------------------------------------------------------------------------
__global__ void cvt_f32_bf16(const float* __restrict__ src,
                             unsigned short* __restrict__ dst, int n) {
    int i = blockIdx.x * blockDim.x + threadIdx.x;
    if (i < n) dst[i] = f2bf(src[i]);
}

// ---------------------------------------------------------------------------
// C[M x N] = A[M x K](bf16) @ W[N x K]^T(bf16) + bias[N](f32)
// One wave per 64x64 C tile. mode 0: bf16 row-major (ld=N)
//                            mode 1: bf16 transposed out[col*ldout + row]
//                            mode 2: f32 row-major (ld=N)
// grid = (N/256, M/64), block = 128 (4 waves, consecutive N tiles)
// __launch_bounds__(128,1): let the 4 waves own the full VGPR file — the
// 64x64 accumulator (128 VGPRs) + fragments must stay in registers, no spill.
// ---------------------------------------------------------------------------
__global__ void __launch_bounds__(128, 1)
gemm_bias_wmma(const unsigned short* __restrict__ A,
               const unsigned short* __restrict__ W,
               const float* __restrict__ bias,
               unsigned short* __restrict__ outb,
               float* __restrict__ outf,
               int K, int N, int ldout, int mode) {
    const int lane   = threadIdx.x & 31;
    const int wave   = threadIdx.x >> 5;
    const int tile_n = (blockIdx.x * 4 + wave) * 64;
    const int tile_m = blockIdx.y * 64;
    const int hi8    = (lane & 16) ? 8 : 0;
    const int col16  = lane & 15;

    v8f acc[4][4] = {};

    for (int kk = 0; kk < K; kk += 32) {
        // prefetch next K-step tiles into cache (global_prefetch_b8)
        if (kk + 32 < K) {
            __builtin_prefetch(A + (size_t)(tile_m + (lane << 1)) * K + kk + 32, 0, 1);
            __builtin_prefetch(W + (size_t)(tile_n + (lane << 1)) * K + kk + 32, 0, 1);
        }

        v16bf bf[4];
#pragma unroll
        for (int j = 0; j < 4; ++j)
            bf[j] = load_b_frag(W, K, lane, tile_n + 16 * j, kk);

        // stream A fragments one at a time: 1 A + 4 B live, acc stays put
#pragma unroll
        for (int i = 0; i < 4; ++i) {
            v16bf a = load_a_frag(A + (size_t)(tile_m + 16 * i) * K + kk, K, lane);
#pragma unroll
            for (int j = 0; j < 4; ++j)
                acc[i][j] = wmma_bf16(a, bf[j], acc[i][j]);
        }
    }

#pragma unroll
    for (int i = 0; i < 4; ++i) {
#pragma unroll
        for (int j = 0; j < 4; ++j) {
            int col = tile_n + 16 * j + col16;
            float b = bias[col];
#pragma unroll
            for (int e = 0; e < 8; ++e) {
                int row = tile_m + 16 * i + e + hi8;
                float v = acc[i][j][e] + b;
                if (mode == 0)       outb[(size_t)row * N + col]     = f2bf(v);
                else if (mode == 1)  outb[(size_t)col * ldout + row] = f2bf(v);
                else                 outf[(size_t)row * N + col]     = v;
            }
        }
    }
}

// ---------------------------------------------------------------------------
// Flash attention: one wave per (head, 16-query tile). Online softmax over
// 32-key blocks; P tile converted C-layout -> A-layout through LDS.
// Qp, Kp: [T, 1024] bf16 row-major. Vt: [1024, T] bf16 (per-head transposed).
// ctx: [T, 1024] bf16.
// ---------------------------------------------------------------------------
__global__ void __launch_bounds__(128, 1)
attn_wmma(const unsigned short* __restrict__ Qp,
          const unsigned short* __restrict__ Kp,
          const unsigned short* __restrict__ Vt,
          unsigned short* __restrict__ ctx) {
    __shared__ unsigned short psh[4][16 * 32];

    const int lane  = threadIdx.x & 31;
    const int wave  = threadIdx.x >> 5;
    const int gw    = blockIdx.x * 4 + wave;
    const int head  = gw >> 8;          // 256 query tiles per head
    const int qtile = gw & 255;
    const int qbase = qtile * 16;
    const int hoff  = head * HEAD_DIM;
    const int hi8   = (lane & 16) ? 8 : 0;
    const int col16 = lane & 15;

    // Q fragments for this 16-row tile (K = head dim, split 0-31 / 32-63)
    const unsigned short* qrow = Qp + (size_t)qbase * D_MODEL + hoff;
    v16bf a0 = load_a_frag(qrow,      D_MODEL, lane);
    v16bf a1 = load_a_frag(qrow + 32, D_MODEL, lane);

    v8f o[4] = {};
    float m8[8], l8[8];
#pragma unroll
    for (int e = 0; e < 8; ++e) { m8[e] = NEG_BIG; l8[e] = 0.0f; }

    const int nkb = ((qbase + 15) >> 5) + 1;   // causal: key blocks of 32
    for (int kb = 0; kb < nkb; ++kb) {
        const int kbase = kb * 32;

        // S tiles: rows = queries, cols = 32 keys of this block
        v8f s0 = {}, s1 = {};
        s0 = wmma_bf16(a0, load_b_frag(Kp, D_MODEL, lane, kbase,      hoff),      s0);
        s0 = wmma_bf16(a1, load_b_frag(Kp, D_MODEL, lane, kbase,      hoff + 32), s0);
        s1 = wmma_bf16(a0, load_b_frag(Kp, D_MODEL, lane, kbase + 16, hoff),      s1);
        s1 = wmma_bf16(a1, load_b_frag(Kp, D_MODEL, lane, kbase + 16, hoff + 32), s1);

#pragma unroll
        for (int e = 0; e < 8; ++e) {
            const int row = qbase + e + hi8;
            const int c0  = kbase + col16;
            const int c1  = c0 + 16;
            float v0 = (c0 <= row) ? s0[e] * ATTN_SCALE : NEG_BIG;
            float v1 = (c1 <= row) ? s1[e] * ATTN_SCALE : NEG_BIG;

            // row max across the 16 lanes of this half-wave group
            float rmax = fmaxf(v0, v1);
            rmax = fmaxf(rmax, __shfl_xor(rmax, 1, 32));
            rmax = fmaxf(rmax, __shfl_xor(rmax, 2, 32));
            rmax = fmaxf(rmax, __shfl_xor(rmax, 4, 32));
            rmax = fmaxf(rmax, __shfl_xor(rmax, 8, 32));

            float mn    = fmaxf(m8[e], rmax);
            float alpha = __expf(m8[e] - mn);
            float p0    = __expf(v0 - mn);
            float p1    = __expf(v1 - mn);

            float rs = p0 + p1;
            rs += __shfl_xor(rs, 1, 32);
            rs += __shfl_xor(rs, 2, 32);
            rs += __shfl_xor(rs, 4, 32);
            rs += __shfl_xor(rs, 8, 32);

            l8[e] = l8[e] * alpha + rs;
            m8[e] = mn;
            o[0][e] *= alpha; o[1][e] *= alpha;
            o[2][e] *= alpha; o[3][e] *= alpha;

            // stash P in LDS (C-layout -> row-major 16x32 bf16)
            const int prow = e + hi8;
            psh[wave][prow * 32 + col16]      = f2bf(p0);
            psh[wave][prow * 32 + 16 + col16] = f2bf(p1);
        }

        // reload P as an A-fragment (16x32, K = 32 keys)
        v16bf pf = load_a_frag(&psh[wave][0], 32, lane);

        // O += P @ V : B-fragment rows = head-dim (Vt rows), K = keys
#pragma unroll
        for (int j = 0; j < 4; ++j)
            o[j] = wmma_bf16(pf, load_b_frag(Vt, T_SEQ, lane,
                                             hoff + 16 * j, kbase), o[j]);
    }

    // normalize and write ctx
#pragma unroll
    for (int j = 0; j < 4; ++j) {
#pragma unroll
        for (int e = 0; e < 8; ++e) {
            int row = qbase + e + hi8;
            int c   = hoff + 16 * j + col16;
            ctx[(size_t)row * D_MODEL + c] = f2bf(o[j][e] / l8[e]);
        }
    }
}

// ---------------------------------------------------------------------------
// Launch: inputs = q,k,v,mask,wq,bq,wk,bk,wv,bv,wo,bo ; out = [4096,1024] f32
// Workspace usage: 64 MB of bf16 staging.
// ---------------------------------------------------------------------------
extern "C" void kernel_launch(void* const* d_in, const int* in_sizes, int n_in,
                              void* d_out, int out_size, void* d_ws, size_t ws_size,
                              hipStream_t stream) {
    (void)in_sizes; (void)n_in; (void)out_size; (void)ws_size;

    const float* q  = (const float*)d_in[0];
    const float* k  = (const float*)d_in[1];
    const float* v  = (const float*)d_in[2];
    // d_in[3] = mask (causal, handled analytically)
    const float* wq = (const float*)d_in[4];
    const float* bq = (const float*)d_in[5];
    const float* wk = (const float*)d_in[6];
    const float* bk = (const float*)d_in[7];
    const float* wv = (const float*)d_in[8];
    const float* bv = (const float*)d_in[9];
    const float* wo = (const float*)d_in[10];
    const float* bo = (const float*)d_in[11];
    float* out = (float*)d_out;

    const size_t E = (size_t)T_SEQ * D_MODEL;     // 4,194,304
    const size_t Wn = (size_t)D_MODEL * D_MODEL;  // 1,048,576
    unsigned short* ws  = (unsigned short*)d_ws;
    unsigned short* qb  = ws;            // bf16 inputs
    unsigned short* kb  = qb + E;
    unsigned short* vb  = kb + E;
    unsigned short* wqb = vb + E;        // bf16 weights
    unsigned short* wkb = wqb + Wn;
    unsigned short* wvb = wkb + Wn;
    unsigned short* wob = wvb + Wn;
    unsigned short* Qp  = wob + Wn;      // projected Q [T,1024]
    unsigned short* Kp  = Qp + E;        // projected K [T,1024]
    unsigned short* Vt  = Kp + E;        // projected V transposed [1024,T]
    unsigned short* ctx = Vt + E;        // attention output [T,1024]

    const int nE = (int)E, nW = (int)Wn;
    cvt_f32_bf16<<<(nE + 255) / 256, 256, 0, stream>>>(q,  qb,  nE);
    cvt_f32_bf16<<<(nE + 255) / 256, 256, 0, stream>>>(k,  kb,  nE);
    cvt_f32_bf16<<<(nE + 255) / 256, 256, 0, stream>>>(v,  vb,  nE);
    cvt_f32_bf16<<<(nW + 255) / 256, 256, 0, stream>>>(wq, wqb, nW);
    cvt_f32_bf16<<<(nW + 255) / 256, 256, 0, stream>>>(wk, wkb, nW);
    cvt_f32_bf16<<<(nW + 255) / 256, 256, 0, stream>>>(wv, wvb, nW);
    cvt_f32_bf16<<<(nW + 255) / 256, 256, 0, stream>>>(wo, wob, nW);

    dim3 gg(D_MODEL / 256, T_SEQ / 64);   // (4, 64)
    dim3 bg(128);
    gemm_bias_wmma<<<gg, bg, 0, stream>>>(qb, wqb, bq, Qp, nullptr,
                                          D_MODEL, D_MODEL, 0, 0);
    gemm_bias_wmma<<<gg, bg, 0, stream>>>(kb, wkb, bk, Kp, nullptr,
                                          D_MODEL, D_MODEL, 0, 0);
    gemm_bias_wmma<<<gg, bg, 0, stream>>>(vb, wvb, bv, Vt, nullptr,
                                          D_MODEL, D_MODEL, T_SEQ, 1);

    attn_wmma<<<(N_HEADS * (T_SEQ / 16)) / 4, 128, 0, stream>>>(Qp, Kp, Vt, ctx);

    gemm_bias_wmma<<<gg, bg, 0, stream>>>(ctx, wob, bo, nullptr, out,
                                          D_MODEL, D_MODEL, 0, 2);
}
